// MultiHeadAttention_2439541424500
// MI455X (gfx1250) — matrix operand order repaired and checked by reading, not compiled, not run on hardware
//
#include <hip/hip_runtime.h>
#include <hip/hip_bf16.h>
#include <math.h>

#define B_    2
#define N_    2048
#define DIM_  1024
#define H_    16
#define DH_   64
#define INNER_ 1024

typedef __attribute__((ext_vector_type(16))) _Float16 v16h;
typedef __attribute__((ext_vector_type(8)))  _Float16 v8h;
typedef __attribute__((ext_vector_type(8)))  float    v8f;

static __device__ __forceinline__ v8f wmma_f16(v16h a, v16h b, v8f c) {
  // D = A(16x32 f16) * B(32x16 f16) + C(16x16 f32)
  return __builtin_amdgcn_wmma_f32_16x16x32_f16(false, a, false, b, (short)0, c, false, false);
}

// A-fragment (16x32 f16), row-major source with leading dim ld.
// lanes 0-15: row=lane,    elems 0-7 = K0..7,  8-15 = K16..23
// lanes16-31: row=lane-16, elems 0-7 = K8..15, 8-15 = K24..31
static __device__ __forceinline__ v16h load_a16x32(const _Float16* base, int ld) {
  const int lane = threadIdx.x & 31;
  const int row  = lane & 15;
  const int kb   = (lane & 16) >> 1;    // 0 or 8
  const _Float16* p = base + (size_t)row * ld + kb;
  v8h lo = *(const v8h*)(p);
  v8h hi = *(const v8h*)(p + 16);
  v16h a;
#pragma unroll
  for (int i = 0; i < 8; ++i) { a[i] = lo[i]; a[8 + i] = hi[i]; }
  return a;
}

// B-fragment (32x16 f16) from a transposed source: row n holds K contiguously.
// lanes 0-15: col=lane, K0..7,K16..23 ; lanes 16-31: col=lane-16, K8..15,K24..31
static __device__ __forceinline__ v16h load_b32x16(const _Float16* bT, int ld) {
  const int lane = threadIdx.x & 31;
  const int col  = lane & 15;
  const int kb   = (lane & 16) >> 1;    // 0 or 8
  const _Float16* p = bT + (size_t)col * ld + kb;
  v8h lo = *(const v8h*)(p);
  v8h hi = *(const v8h*)(p + 16);
  v16h b;
#pragma unroll
  for (int i = 0; i < 8; ++i) { b[i] = lo[i]; b[8 + i] = hi[i]; }
  return b;
}

// ---------------- Kernel 0: f32 -> f16 conversion (+ weight transpose) -----
__global__ __launch_bounds__(256) void convert_kernel(
    const float* __restrict__ x, const float* __restrict__ wq,
    const float* __restrict__ wk, const float* __restrict__ wv,
    const float* __restrict__ wo,
    _Float16* __restrict__ xh, _Float16* __restrict__ wqT,
    _Float16* __restrict__ wkT, _Float16* __restrict__ wvT,
    _Float16* __restrict__ woT) {
  const int XN = B_ * N_ * DIM_;      // 4,194,304
  const int WN = DIM_ * INNER_;       // 1,048,576
  int i = blockIdx.x * 256 + threadIdx.x;
  if (i < XN) { xh[i] = (_Float16)x[i]; return; }
  int j = i - XN;
  int w = j / WN;
  int t = j - w * WN;
  int n = t >> 10;
  int k = t & 1023;
  const float* src = (w == 0) ? wq : (w == 1) ? wk : (w == 2) ? wv : wo;
  _Float16* dst = (w == 0) ? wqT : (w == 1) ? wkT : (w == 2) ? wvT : woT;
  dst[t] = (_Float16)src[k * 1024 + n];   // dstT[n][k] = src[k][n]
}

// ---------------- Kernel 1: QKV projection (WMMA) --------------------------
__global__ __launch_bounds__(256) void qkv_kernel(
    const _Float16* __restrict__ xh, const _Float16* __restrict__ wqT,
    const _Float16* __restrict__ wkT, const _Float16* __restrict__ wvT,
    float* __restrict__ qf, float* __restrict__ kst, float* __restrict__ vst,
    _Float16* __restrict__ vTh) {
  const int warp = threadIdx.x >> 5, lane = threadIdx.x & 31;
  const int wid = blockIdx.x * 8 + warp;   // 4096 waves
  const int mt = wid >> 4;                 // 0..255 row tiles (16 rows each)
  const int nt = wid & 15;                 // 0..15 col tiles (64 cols) == head
  const _Float16* abase = xh + (size_t)mt * 16 * DIM_;
  v8f aq[4] = {}, ak[4] = {}, av[4] = {};
  for (int k0 = 0; k0 < DIM_; k0 += 32) {
    v16h a = load_a16x32(abase + k0, DIM_);
#pragma unroll
    for (int ct = 0; ct < 4; ++ct) {
      size_t cb = (size_t)(nt * 64 + ct * 16) * DIM_ + k0;
      aq[ct] = wmma_f16(a, load_b32x16(wqT + cb, DIM_), aq[ct]);
      ak[ct] = wmma_f16(a, load_b32x16(wkT + cb, DIM_), ak[ct]);
      av[ct] = wmma_f16(a, load_b32x16(wvT + cb, DIM_), av[ct]);
    }
  }
  const int h = nt;
  const int row0 = mt * 16;
  const int b = row0 >> 11;                // /N_
  const int n0 = row0 & (N_ - 1);
  const int roff = (lane & 16) >> 1;       // 0 or 8
#pragma unroll
  for (int ct = 0; ct < 4; ++ct) {
    const int dh = ct * 16 + (lane & 15);
#pragma unroll
    for (int r = 0; r < 8; ++r) {
      const int n = n0 + r + roff;
      size_t idx = ((size_t)(b * H_ + h) * N_ + n) * DH_ + dh;   // (B,H,N,DH)
      qf[idx] = aq[ct][r];
      kst[idx] = ak[ct][r];                 // pre-RoPE K staged in d_out
      vst[idx] = av[ct][r];                 // V output (f32)
      vTh[((size_t)(b * H_ + h) * DH_ + dh) * N_ + n] = (_Float16)av[ct][r];
    }
  }
}

// ---------------- Kernel 2: RoPE (+scale) ----------------------------------
__global__ __launch_bounds__(256) void rope_kernel(
    const float* __restrict__ qf, float* __restrict__ kio,
    _Float16* __restrict__ qh, _Float16* __restrict__ kh) {
  const int t = blockIdx.x * 256 + threadIdx.x;   // B*H*N = 65536 rows
  const int pos = t & (N_ - 1);
  const float* qrow = qf + (size_t)t * DH_;
  float* krow = kio + (size_t)t * DH_;
  _Float16* qhr = qh + (size_t)t * DH_;
  _Float16* khr = kh + (size_t)t * DH_;
  float qv[DH_], kv[DH_];
#pragma unroll
  for (int d = 0; d < DH_; ++d) { qv[d] = qrow[d]; kv[d] = krow[d]; }
  const float LN = 9.210340371976184f;  // ln(10000)
#pragma unroll
  for (int j = 0; j < 32; ++j) {
    float invf = expf(-(float)(2 * j) * (1.0f / 64.0f) * LN);
    float ang = (float)pos * invf;
    float c = cosf(ang), s = sinf(ang);
    // rot = concat(-x[1::2], x[0::2])
    float q0 = qv[j] * c - qv[2 * j + 1] * s;
    float q1 = qv[j + 32] * c + qv[2 * j] * s;
    float k0 = kv[j] * c - kv[2 * j + 1] * s;
    float k1 = kv[j + 32] * c + kv[2 * j] * s;
    krow[j] = k0; krow[j + 32] = k1;                 // roped K (f32 output)
    khr[j] = (_Float16)k0; khr[j + 32] = (_Float16)k1;
    qhr[j] = (_Float16)(q0 * 0.125f);                // fold in DH^-0.5
    qhr[j + 32] = (_Float16)(q1 * 0.125f);
  }
}

// ---------------- Kernel 3: causal flash attention (WMMA) ------------------
__global__ __launch_bounds__(128) void attn_kernel(
    const _Float16* __restrict__ qh, const _Float16* __restrict__ kh,
    const _Float16* __restrict__ vT, _Float16* __restrict__ attnh) {
  __shared__ __align__(16) _Float16 plds[4][16 * 32];   // per-wave P tile
  const int warp = threadIdx.x >> 5, lane = threadIdx.x & 31;
  const int wid = blockIdx.x * 4 + warp;    // 4096 waves
  const int qt = wid & 127;
  const int bh = wid >> 7;                  // 0..31
  const int b = bh >> 4, h = bh & 15;
  const _Float16* qbh = qh + (size_t)bh * N_ * DH_;
  const _Float16* kbh = kh + (size_t)bh * N_ * DH_;
  const _Float16* vbh = vT + (size_t)bh * DH_ * N_;
  const int qb = qt * 16;
  const v16h aQ0 = load_a16x32(qbh + (size_t)qb * DH_, DH_);
  const v16h aQ1 = load_a16x32(qbh + (size_t)qb * DH_ + 32, DH_);
  v8f o0 = {}, o1 = {}, o2 = {}, o3 = {};
  float mr[8], lr[8];
#pragma unroll
  for (int r = 0; r < 8; ++r) { mr[r] = -3.0e38f; lr[r] = 0.0f; }
  const int roff = (lane & 16) >> 1;        // 0 or 8 (row split of C layout)
  const int nblk = (qb + 16 + 31) >> 5;     // 32-key blocks up to diagonal
  _Float16* pw = &plds[warp][0];
  for (int blk = 0; blk < nblk; ++blk) {
    const int k0 = blk * 32;
    v8f s0 = {}, s1 = {};
    s0 = wmma_f16(aQ0, load_b32x16(kbh + (size_t)k0 * DH_, DH_), s0);
    s0 = wmma_f16(aQ1, load_b32x16(kbh + (size_t)k0 * DH_ + 32, DH_), s0);
    s1 = wmma_f16(aQ0, load_b32x16(kbh + (size_t)(k0 + 16) * DH_, DH_), s1);
    s1 = wmma_f16(aQ1, load_b32x16(kbh + (size_t)(k0 + 16) * DH_ + 32, DH_), s1);
    if (k0 + 31 > qb) {                      // diagonal block: causal mask
      const int key0 = k0 + (lane & 15);
#pragma unroll
      for (int r = 0; r < 8; ++r) {
        const int qrow = qb + r + roff;
        if (key0 > qrow)      s0[r] = -3.0e38f;
        if (key0 + 16 > qrow) s1[r] = -3.0e38f;
      }
    }
#pragma unroll
    for (int r = 0; r < 8; ++r) {
      float mx = fmaxf(s0[r], s1[r]);
      mx = fmaxf(mx, __shfl_xor(mx, 1, 32));
      mx = fmaxf(mx, __shfl_xor(mx, 2, 32));
      mx = fmaxf(mx, __shfl_xor(mx, 4, 32));
      mx = fmaxf(mx, __shfl_xor(mx, 8, 32));
      const float mnew = fmaxf(mr[r], mx);
      const float alpha = __expf(mr[r] - mnew);
      const float p0 = __expf(s0[r] - mnew);
      const float p1 = __expf(s1[r] - mnew);
      float ps = p0 + p1;
      ps += __shfl_xor(ps, 1, 32);
      ps += __shfl_xor(ps, 2, 32);
      ps += __shfl_xor(ps, 4, 32);
      ps += __shfl_xor(ps, 8, 32);
      lr[r] = lr[r] * alpha + ps;
      mr[r] = mnew;
      o0[r] *= alpha; o1[r] *= alpha; o2[r] *= alpha; o3[r] *= alpha;
      const int prow = r + roff;            // C-layout row owned by this lane
      pw[prow * 32 + (lane & 15)] = (_Float16)p0;
      pw[prow * 32 + 16 + (lane & 15)] = (_Float16)p1;
    }
    // same-wave LDS ops are in-order: re-read P in A-fragment layout
    const v16h aP = load_a16x32(pw, 32);
    o0 = wmma_f16(aP, load_b32x16(vbh + k0, N_), o0);
    o1 = wmma_f16(aP, load_b32x16(vbh + (size_t)16 * N_ + k0, N_), o1);
    o2 = wmma_f16(aP, load_b32x16(vbh + (size_t)32 * N_ + k0, N_), o2);
    o3 = wmma_f16(aP, load_b32x16(vbh + (size_t)48 * N_ + k0, N_), o3);
  }
#pragma unroll
  for (int r = 0; r < 8; ++r) {
    const float inv = 1.0f / lr[r];
    o0[r] *= inv; o1[r] *= inv; o2[r] *= inv; o3[r] *= inv;
  }
  const size_t outbase = (size_t)b * N_ * INNER_;
#pragma unroll
  for (int r = 0; r < 8; ++r) {
    const int row = qb + r + roff;
    size_t rb = outbase + (size_t)row * INNER_ + h * 64 + (lane & 15);
    attnh[rb]      = (_Float16)o0[r];
    attnh[rb + 16] = (_Float16)o1[r];
    attnh[rb + 32] = (_Float16)o2[r];
    attnh[rb + 48] = (_Float16)o3[r];
  }
}

// ---------------- Kernel 4: output projection + bias -----------------------
__global__ __launch_bounds__(256) void out_kernel(
    const _Float16* __restrict__ ah, const _Float16* __restrict__ woT,
    const float* __restrict__ bo, float* __restrict__ out) {
  const int warp = threadIdx.x >> 5, lane = threadIdx.x & 31;
  const int wid = blockIdx.x * 8 + warp;
  const int mt = wid >> 4;
  const int nt = wid & 15;
  const _Float16* abase = ah + (size_t)mt * 16 * INNER_;
  v8f acc[4] = {};
  for (int k0 = 0; k0 < INNER_; k0 += 32) {
    v16h a = load_a16x32(abase + k0, INNER_);
#pragma unroll
    for (int ct = 0; ct < 4; ++ct) {
      size_t cb = (size_t)(nt * 64 + ct * 16) * INNER_ + k0;
      acc[ct] = wmma_f16(a, load_b32x16(woT + cb, INNER_), acc[ct]);
    }
  }
  const int roff = (lane & 16) >> 1;
#pragma unroll
  for (int ct = 0; ct < 4; ++ct) {
    const int col = nt * 64 + ct * 16 + (lane & 15);
    const float bias = bo[col];
#pragma unroll
    for (int r = 0; r < 8; ++r) {
      const int row = mt * 16 + r + roff;
      out[(size_t)row * DIM_ + col] = acc[ct][r] + bias;
    }
  }
}

extern "C" void kernel_launch(void* const* d_in, const int* in_sizes, int n_in,
                              void* d_out, int out_size, void* d_ws, size_t ws_size,
                              hipStream_t stream) {
  (void)in_sizes; (void)n_in; (void)out_size; (void)ws_size;
  const float* x  = (const float*)d_in[0];
  const float* Wq = (const float*)d_in[1];
  const float* Wk = (const float*)d_in[2];
  const float* Wv = (const float*)d_in[3];
  const float* Wo = (const float*)d_in[4];
  const float* bo = (const float*)d_in[5];

  float* out  = (float*)d_out;                         // (B,N,DIM)
  float* kout = out + (size_t)B_ * N_ * DIM_;          // (B,H,N,DH)
  float* vout = kout + (size_t)B_ * H_ * N_ * DH_;     // (B,H,N,DH)

  char* w = (char*)d_ws;                               // 64 MiB total
  _Float16* xh  = (_Float16*)w; w += (size_t)B_ * N_ * DIM_ * 2;
  _Float16* wqT = (_Float16*)w; w += (size_t)DIM_ * INNER_ * 2;
  _Float16* wkT = (_Float16*)w; w += (size_t)DIM_ * INNER_ * 2;
  _Float16* wvT = (_Float16*)w; w += (size_t)DIM_ * INNER_ * 2;
  _Float16* woT = (_Float16*)w; w += (size_t)INNER_ * DIM_ * 2;
  float*    qf  = (float*)w;    w += (size_t)B_ * H_ * N_ * DH_ * 4;
  _Float16* qh  = (_Float16*)w; w += (size_t)B_ * H_ * N_ * DH_ * 2;
  _Float16* kh  = (_Float16*)w; w += (size_t)B_ * H_ * N_ * DH_ * 2;
  _Float16* vTh = (_Float16*)w; w += (size_t)B_ * H_ * N_ * DH_ * 2;
  _Float16* ah  = (_Float16*)w; w += (size_t)B_ * N_ * INNER_ * 2;

  convert_kernel<<<32768, 256, 0, stream>>>(x, Wq, Wk, Wv, Wo,
                                            xh, wqT, wkT, wvT, woT);
  qkv_kernel<<<512, 256, 0, stream>>>(xh, wqT, wkT, wvT, qf, kout, vout, vTh);
  rope_kernel<<<256, 256, 0, stream>>>(qf, kout, qh, kh);
  attn_kernel<<<1024, 128, 0, stream>>>(qh, kh, vTh, ah);
  out_kernel<<<512, 256, 0, stream>>>(ah, woT, bo, out);
}
